// LstmCharGen_77799037599830
// MI455X (gfx1250) — matrix operand order repaired and checked by reading, not compiled, not run on hardware
//
#include <hip/hip_runtime.h>
#include <hip/hip_bf16.h>

typedef __bf16 bf16;
typedef __attribute__((ext_vector_type(16))) __bf16 bf16x16;
typedef __attribute__((ext_vector_type(8)))  __bf16 bf16x8;
typedef __attribute__((ext_vector_type(8)))  float  floatx8;

// Problem constants (V,E,H,L,B,T from the reference)
#define VOC   10000
#define EMB   512
#define HID   1024
#define GATES 4096   // 4*H
#define BB    32
#define TT    512
#define MROWS (TT * BB)   // 16384 rows in (t,b) order

// ---------------------------------------------------------------------------
// helpers
// ---------------------------------------------------------------------------
__device__ __forceinline__ bf16 f2bf(float f) {
  unsigned u = __builtin_bit_cast(unsigned, f);
  u += 0x7FFFu + ((u >> 16) & 1u);               // round-to-nearest-even
  unsigned short h = (unsigned short)(u >> 16);
  return __builtin_bit_cast(bf16, h);
}

__device__ __forceinline__ float sigmoidf_fast(float x) {
  return 1.0f / (1.0f + __expf(-x));
}

// A-tile 16x32 bf16, ISA register layout:
//  lanes 0-15 : row = lane,    K = {k0..k0+7,  k0+16..k0+23}
//  lanes 16-31: row = lane-16, K = {k0+8..k0+15, k0+24..k0+31}
__device__ __forceinline__ bf16x16 load_a16x32(const bf16* __restrict__ A, int ld,
                                               int m0, int k0, int lane) {
  const int r  = m0 + (lane & 15);
  const int ks = k0 + ((lane & 16) ? 8 : 0);
  const bf16* p = A + (size_t)r * ld + ks;
  bf16x8 lo = *reinterpret_cast<const bf16x8*>(p);
  bf16x8 hi = *reinterpret_cast<const bf16x8*>(p + 16);
  bf16x16 out;
#pragma unroll
  for (int i = 0; i < 8; ++i) { out[i] = lo[i]; out[i + 8] = hi[i]; }
  return out;
}

// B-tile 32x16 (KxN) where B = W^T and W is stored [N,K] row-major:
//  lanes 0-15 : col = n0+lane,    K = {k0..k0+7,  k0+16..k0+23}
//  lanes 16-31: col = n0+lane-16, K = {k0+8..k0+15, k0+24..k0+31}
__device__ __forceinline__ bf16x16 load_b32x16_wT(const bf16* __restrict__ W, int ld,
                                                  int n0, int k0, int lane) {
  const int n  = n0 + (lane & 15);
  const int ks = k0 + ((lane & 16) ? 8 : 0);
  const bf16* p = W + (size_t)n * ld + ks;
  bf16x8 lo = *reinterpret_cast<const bf16x8*>(p);
  bf16x8 hi = *reinterpret_cast<const bf16x8*>(p + 16);
  bf16x16 out;
#pragma unroll
  for (int i = 0; i < 8; ++i) { out[i] = lo[i]; out[i + 8] = hi[i]; }
  return out;
}

__device__ __forceinline__ floatx8 wmma_bf16(bf16x16 a, bf16x16 b, floatx8 c) {
  return __builtin_amdgcn_wmma_f32_16x16x32_bf16(false, a, false, b, (short)0, c,
                                                 false, false);
}

// ---------------------------------------------------------------------------
// elementwise utility kernels
// ---------------------------------------------------------------------------
__global__ void cvt_f32_bf16_kernel(const float* __restrict__ src,
                                    bf16* __restrict__ dst, long n) {
  long i = (long)blockIdx.x * blockDim.x + threadIdx.x;
  long stride = (long)gridDim.x * blockDim.x;
  for (; i < n; i += stride) dst[i] = f2bf(src[i]);
}

__global__ void zero_u32_kernel(unsigned* __restrict__ p, long n) {
  long i = (long)blockIdx.x * blockDim.x + threadIdx.x;
  long stride = (long)gridDim.x * blockDim.x;
  for (; i < n; i += stride) p[i] = 0u;
}

// embedding gather + bf16 convert: X0[(t*B+b), :] = bf16(embedding[x[b,t], :])
__global__ void gather_embed_kernel(const int* __restrict__ x,
                                    const float* __restrict__ emb,
                                    bf16* __restrict__ X0) {
  const int tb = blockIdx.x;           // t*B + b
  const int t = tb >> 5;
  const int b = tb & 31;
  const int row = x[b * TT + t];
  const float* src = emb + (size_t)row * EMB;
  bf16* dst = X0 + (size_t)tb * EMB;
  for (int e = threadIdx.x; e < EMB; e += blockDim.x) dst[e] = f2bf(src[e]);
}

// pack final states behind the logits: d_out = [logits | h_state(2,B,H) | c_state(2,B,H)]
__global__ void finalize_states_kernel(const float* __restrict__ h0,
                                       const float* __restrict__ h1,
                                       const float* __restrict__ c0,
                                       const float* __restrict__ c1,
                                       float* __restrict__ out) {
  int i = blockIdx.x * blockDim.x + threadIdx.x;     // 0 .. 4*32*1024-1
  if (i >= 4 * BB * HID) return;
  const int which = i >> 15;                          // 32*1024 = 32768 per chunk
  const int off = i & 32767;
  const float* src = (which == 0) ? h0 : (which == 1) ? h1 : (which == 2) ? c0 : c1;
  out[i] = src[off];
}

// ---------------------------------------------------------------------------
// Big batched WMMA GEMM:  D[M,N] = A[M,K](bf16) * W[N,K]^T(bf16) + bias[N]
// Workgroup tile 128x128, 8 waves, each wave 64x32 (4x2 subtiles of 16x16).
// swizzleBT: remap output row r=(t*B+b) -> b*T+t   (for the classifier)
// ---------------------------------------------------------------------------
__global__ __launch_bounds__(256) void wmma_gemm_bias_kernel(
    const bf16* __restrict__ A, int lda,
    const bf16* __restrict__ W, int ldw,
    const float* __restrict__ bias,
    float* __restrict__ D, long ldd,
    int M, int N, int K, int swizzleBT) {
  const int lane = threadIdx.x & 31;
  const int wid  = threadIdx.x >> 5;                   // 0..7
  const int m0 = blockIdx.x * 128 + (wid & 1) * 64;
  const int n0 = blockIdx.y * 128 + (wid >> 1) * 32;

  floatx8 acc[4][2] = {};

  for (int k0 = 0; k0 < K; k0 += 32) {
    if (k0 + 32 < K) {
      __builtin_prefetch(A + (size_t)m0 * lda + (k0 + 32), 0, 1);
      __builtin_prefetch(W + (size_t)n0 * ldw + (k0 + 32), 0, 1);
    }
    bf16x16 bt[2];
#pragma unroll
    for (int j = 0; j < 2; ++j) {
      int nn = n0 + 16 * j;
      if (nn > N - 16) nn = N - 16;                    // clamp (N multiple of 16)
      bt[j] = load_b32x16_wT(W, ldw, nn, k0, lane);
    }
#pragma unroll
    for (int i = 0; i < 4; ++i) {
      bf16x16 at = load_a16x32(A, lda, m0 + 16 * i, k0, lane);
#pragma unroll
      for (int j = 0; j < 2; ++j)
        acc[i][j] = wmma_bf16(at, bt[j], acc[i][j]);
    }
  }

  // bias + store (C/D layout: VGPR rr -> row rbase+rr, col = lane&15 of subtile)
#pragma unroll
  for (int i = 0; i < 4; ++i) {
    const int rbase = m0 + 16 * i + ((lane & 16) ? 8 : 0);
#pragma unroll
    for (int j = 0; j < 2; ++j) {
      const int nn = n0 + 16 * j;
      if (nn >= N) continue;
      const int col = nn + (lane & 15);
      const float bv = bias[col];
#pragma unroll
      for (int rr = 0; rr < 8; ++rr) {
        const int r = rbase + rr;
        const long orow = swizzleBT ? ((long)(r & (BB - 1)) * TT + (r >> 5)) : (long)r;
        D[orow * ldd + col] = acc[i][j][rr] + bv;
      }
    }
  }
}

// ---------------------------------------------------------------------------
// One LSTM timestep. Grid: 32 workgroups (one per 32-column hidden slice),
// 128 threads = 4 waves, wave g computes gate g's 32x32 strip of
//   z = xW_t + h_{t-1} * W_hh^T
// h_{t-1} (32x1024 bf16, 64 KB) staged in LDS; z exchanged through LDS so the
// c/h pointwise update is workgroup-local. Dynamic LDS = 64K + 16K = 80 KB.
// ---------------------------------------------------------------------------
__global__ __launch_bounds__(128) void lstm_step_kernel(
    const float* __restrict__ XW,        // [T*B, 4H] precomputed x-projection (+bias)
    const bf16*  __restrict__ Whh,       // [4H, H] bf16
    const bf16*  __restrict__ h_prev,    // [B, H] bf16
    bf16*        __restrict__ h_next,    // [B, H] bf16
    float*       __restrict__ c,         // [B, H] f32 (updated in place)
    bf16*        __restrict__ Hseq,      // [T*B, H] bf16 (layer output sequence)
    float*       __restrict__ h_state,   // [B, H] f32 (final-h snapshot)
    int t) {
  extern __shared__ char smem[];
  bf16*  sh = reinterpret_cast<bf16*>(smem);                 // [32][1024] bf16
  float* sz = reinterpret_cast<float*>(smem + BB * HID * 2); // [4][32][32] f32

  // stage h_prev into LDS (32*1024*2 B = 4096 uint4)
  {
    const uint4* src = reinterpret_cast<const uint4*>(h_prev);
    uint4* dst = reinterpret_cast<uint4*>(sh);
    for (int i = threadIdx.x; i < (BB * HID * 2) / 16; i += 128) dst[i] = src[i];
  }
  __syncthreads();

  const int lane = threadIdx.x & 31;
  const int gate = threadIdx.x >> 5;          // 0:i 1:f 2:g 3:o
  const int j0 = blockIdx.x * 32;             // hidden-column slice
  const int nbase = gate * HID + j0;          // column base in 4H gate dimension

  // accumulators preloaded with xW_t tile -> WMMA computes z directly
  floatx8 acc[2][2];
#pragma unroll
  for (int i = 0; i < 2; ++i) {
    const int rb = 16 * i + ((lane & 16) ? 8 : 0);
#pragma unroll
    for (int j = 0; j < 2; ++j) {
      const int col = nbase + 16 * j + (lane & 15);
#pragma unroll
      for (int rr = 0; rr < 8; ++rr)
        acc[i][j][rr] = XW[(size_t)(t * BB + rb + rr) * GATES + col];
    }
  }

  for (int k0 = 0; k0 < HID; k0 += 32) {
    bf16x16 at[2];
#pragma unroll
    for (int i = 0; i < 2; ++i) at[i] = load_a16x32(sh, HID, 16 * i, k0, lane);
    bf16x16 bt[2];
#pragma unroll
    for (int j = 0; j < 2; ++j) bt[j] = load_b32x16_wT(Whh, HID, nbase + 16 * j, k0, lane);
#pragma unroll
    for (int i = 0; i < 2; ++i)
#pragma unroll
      for (int j = 0; j < 2; ++j)
        acc[i][j] = wmma_bf16(at[i], bt[j], acc[i][j]);
  }

  // park z strips in LDS, gate-major
#pragma unroll
  for (int i = 0; i < 2; ++i) {
    const int rb = 16 * i + ((lane & 16) ? 8 : 0);
#pragma unroll
    for (int j = 0; j < 2; ++j) {
      const int cl = 16 * j + (lane & 15);
#pragma unroll
      for (int rr = 0; rr < 8; ++rr)
        sz[gate * (BB * 32) + (rb + rr) * 32 + cl] = acc[i][j][rr];
    }
  }
  __syncthreads();

  // pointwise gates + state update for this 32-column slice
  for (int e = threadIdx.x; e < BB * 32; e += 128) {
    const int row = e >> 5;                   // batch index
    const int cl  = e & 31;
    const float zi = sz[e];
    const float zf = sz[1024 + e];
    const float zg = sz[2048 + e];
    const float zo = sz[3072 + e];
    const float ig = sigmoidf_fast(zi);
    const float fg = sigmoidf_fast(zf);
    const float og = sigmoidf_fast(zo);
    const float gg = tanhf(zg);
    const size_t idx = (size_t)row * HID + j0 + cl;
    const float cn = fg * c[idx] + ig * gg;
    const float hn = og * tanhf(cn);
    c[idx] = cn;
    h_next[idx] = f2bf(hn);
    h_state[idx] = hn;
    Hseq[(size_t)(t * BB) * HID + idx] = f2bf(hn);
  }
}

// ---------------------------------------------------------------------------
// host: workspace layout + launch sequence
// ---------------------------------------------------------------------------
extern "C" void kernel_launch(void* const* d_in, const int* in_sizes, int n_in,
                              void* d_out, int out_size, void* d_ws, size_t ws_size,
                              hipStream_t stream) {
  (void)in_sizes; (void)n_in; (void)out_size; (void)ws_size;

  const int*   x      = (const int*)  d_in[0];
  const float* emb    = (const float*)d_in[1];
  const float* W_ih0  = (const float*)d_in[2];
  const float* W_hh0  = (const float*)d_in[3];
  const float* b0     = (const float*)d_in[4];
  const float* W_ih1  = (const float*)d_in[5];
  const float* W_hh1  = (const float*)d_in[6];
  const float* b1     = (const float*)d_in[7];
  const float* W_cls  = (const float*)d_in[8];
  const float* b_cls  = (const float*)d_in[9];
  float* out = (float*)d_out;

  char* ws = (char*)d_ws;
  size_t off = 0;
  auto carve = [&](size_t bytes) { char* p = ws + off; off += (bytes + 255) & ~(size_t)255; return p; };

  bf16* wih0b = (bf16*)carve((size_t)GATES * EMB * 2);
  bf16* whh0b = (bf16*)carve((size_t)GATES * HID * 2);
  bf16* wih1b = (bf16*)carve((size_t)GATES * HID * 2);
  bf16* whh1b = (bf16*)carve((size_t)GATES * HID * 2);
  bf16* wclsb = (bf16*)carve((size_t)VOC * HID * 2);
  bf16* X0    = (bf16*)carve((size_t)MROWS * EMB * 2);
  bf16* Hseq  = (bf16*)carve((size_t)MROWS * HID * 2);
  float* XW   = (float*)carve((size_t)MROWS * GATES * 4);
  bf16* hA    = (bf16*)carve((size_t)BB * HID * 2);
  bf16* hB    = (bf16*)carve((size_t)BB * HID * 2);
  float* c0   = (float*)carve((size_t)BB * HID * 4);
  float* c1   = (float*)carve((size_t)BB * HID * 4);
  float* hs0  = (float*)carve((size_t)BB * HID * 4);
  float* hs1  = (float*)carve((size_t)BB * HID * 4);

  // 1) weights -> bf16
  cvt_f32_bf16_kernel<<<4096, 256, 0, stream>>>(W_ih0, wih0b, (long)GATES * EMB);
  cvt_f32_bf16_kernel<<<4096, 256, 0, stream>>>(W_hh0, whh0b, (long)GATES * HID);
  cvt_f32_bf16_kernel<<<4096, 256, 0, stream>>>(W_ih1, wih1b, (long)GATES * HID);
  cvt_f32_bf16_kernel<<<4096, 256, 0, stream>>>(W_hh1, whh1b, (long)GATES * HID);
  cvt_f32_bf16_kernel<<<4096, 256, 0, stream>>>(W_cls, wclsb, (long)VOC * HID);

  // 2) embedding gather -> X0 in (t,b) row order
  gather_embed_kernel<<<MROWS, 128, 0, stream>>>(x, emb, X0);

  // 3) layer 0 input projection: XW = X0 * W_ih0^T + b0
  wmma_gemm_bias_kernel<<<dim3(MROWS / 128, GATES / 128), 256, 0, stream>>>(
      X0, EMB, wih0b, EMB, b0, XW, GATES, MROWS, GATES, EMB, 0);

  // 4) layer 0 recurrence
  zero_u32_kernel<<<64, 256, 0, stream>>>((unsigned*)hA, (long)BB * HID * 2 / 4);
  zero_u32_kernel<<<64, 256, 0, stream>>>((unsigned*)c0, (long)BB * HID);
  for (int t = 0; t < TT; ++t) {
    bf16* hp = (t & 1) ? hB : hA;
    bf16* hn = (t & 1) ? hA : hB;
    lstm_step_kernel<<<HID / 32, 128, 81920, stream>>>(XW, whh0b, hp, hn, c0, Hseq, hs0, t);
  }

  // 5) layer 1 input projection: XW = Hseq * W_ih1^T + b1
  wmma_gemm_bias_kernel<<<dim3(MROWS / 128, GATES / 128), 256, 0, stream>>>(
      Hseq, HID, wih1b, HID, b1, XW, GATES, MROWS, GATES, HID, 0);

  // 6) layer 1 recurrence (Hseq reused for layer-1 outputs; XW already consumed)
  zero_u32_kernel<<<64, 256, 0, stream>>>((unsigned*)hA, (long)BB * HID * 2 / 4);
  zero_u32_kernel<<<64, 256, 0, stream>>>((unsigned*)c1, (long)BB * HID);
  for (int t = 0; t < TT; ++t) {
    bf16* hp = (t & 1) ? hB : hA;
    bf16* hn = (t & 1) ? hA : hB;
    lstm_step_kernel<<<HID / 32, 128, 81920, stream>>>(XW, whh1b, hp, hn, c1, Hseq, hs1, t);
  }

  // 7) classifier: logits[b,t,:] = Hseq[(t,b),:] * W_cls^T + b_cls  (row swizzle)
  wmma_gemm_bias_kernel<<<dim3(MROWS / 128, (VOC + 127) / 128), 256, 0, stream>>>(
      Hseq, HID, wclsb, HID, b_cls, out, VOC, MROWS, VOC, HID, 1);

  // 8) final h/c states behind the logits
  finalize_states_kernel<<<(4 * BB * HID + 255) / 256, 256, 0, stream>>>(
      hs0, hs1, c0, c1, out + (size_t)BB * TT * VOC);
}
